// AttentionBlock_19902878449746
// MI455X (gfx1250) — matrix operand order repaired and checked by reading, not compiled, not run on hardware
//
#include <hip/hip_runtime.h>

#define DEV_INLINE __device__ __forceinline__

typedef __attribute__((ext_vector_type(16))) __bf16 v16bf;
typedef __attribute__((ext_vector_type(8)))  float  v8f;
typedef __attribute__((ext_vector_type(4)))  unsigned int u32x4;
typedef __attribute__((ext_vector_type(8)))  int i32x8;
typedef __attribute__((ext_vector_type(4)))  int i32x4;

union Frag16 { u32x4 u[2]; v16bf v; };

constexpr int Cc = 256;    // channels
constexpr int Np = 4096;   // H*W
constexpr int Bb = 8;      // batch
constexpr int KB = 128;    // key-block size
constexpr int KPAD = 264;  // K tile LDS row stride (elems): 512B + 16B skew
constexpr int VPAD = 136;  // V tile LDS row stride (elems): 256B + 16B skew
constexpr float SCALE = 0.0625f;             // C^-0.5
constexpr float LOG2E = 1.4426950408889634f;

// Load a 16-element bf16 WMMA fragment as two 16B chunks.
DEV_INLINE v16bf load_frag(const __bf16* p0, const __bf16* p1) {
  Frag16 f;
  f.u[0] = *reinterpret_cast<const u32x4*>(p0);
  f.u[1] = *reinterpret_cast<const u32x4*>(p1);
  return f.v;
}

DEV_INLINE v8f wmma_bf16(v16bf a, v16bf b, v8f c) {
  // D(f32 16x16) = A(bf16 16x32) * B(bf16 32x16) + C
  return __builtin_amdgcn_wmma_f32_16x16x32_bf16(false, a, false, b,
                                                 (short)0, c, false, false);
}

// ---------------------------------------------------------------------------
// Generic 2D TDM tile load: rows x cols (bf16), global row stride gstride,
// with LDS row padding (pad_interval/pad_amount preset in ctl0 per tile shape).
// D# fields packed per CDNA5 ISA.
// ---------------------------------------------------------------------------
DEV_INLINE void tdm_load_tile(const __bf16* gsrc, unsigned lds_addr,
                              unsigned ctl0, unsigned tdim0, unsigned tdim1,
                              unsigned tile0, unsigned tile1, unsigned stride0) {
  unsigned long long ga = (unsigned long long)(const void*)gsrc;
  u32x4 g0;
  g0[0] = 1u;                                   // count=1 (valid user descriptor)
  g0[1] = lds_addr;                             // lds_addr [63:32]
  g0[2] = (unsigned)(ga & 0xffffffffu);         // global_addr[31:0]
  g0[3] = (unsigned)((ga >> 32) & 0x01ffffffu)  // global_addr[56:32]
          | (2u << 30);                         // type = 2 ("image")
  i32x8 g1;
  g1[0] = (int)ctl0;                  // data_size=2B | pad ctl
  g1[1] = (int)(tdim0 << 16);         // tensor_dim0 low half -> bits 79:48
  g1[2] = (int)((tdim0 >> 16) | (tdim1 << 16));   // dim0 hi | dim1 lo
  g1[3] = (int)((tdim1 >> 16) | (tile0 << 16));   // dim1 hi | tile_dim0
  g1[4] = (int)tile1;                 // tile_dim1 | tile_dim2=0
  g1[5] = (int)stride0;               // tensor_dim0_stride (48b, low dword)
  g1[6] = 0;
  g1[7] = 0;                          // tensor_dim1_stride unused (2D tile)
  i32x4 gz = {0, 0, 0, 0};
#if defined(__clang_major__) && (__clang_major__ >= 23)
  i32x8 gz8 = {0, 0, 0, 0, 0, 0, 0, 0};
  __builtin_amdgcn_tensor_load_to_lds(g0, g1, gz, gz, gz8, 0);
#else
  __builtin_amdgcn_tensor_load_to_lds(g0, g1, gz, gz, 0);
#endif
}

// ctl0 fields: [17:16]=data_size(1->2B) [20]=pad_en [24:22]=pad_interval [31:25]=pad_amount
// K tile: rows of 256 elems = 128 DWORDs (interval code 6), pad 4 DWORDs (code 3)
constexpr unsigned CTL_K = 0x00010000u | (1u << 20) | (6u << 22) | (3u << 25);
// V tile: rows of 128 elems = 64 DWORDs (interval code 5), pad 4 DWORDs (code 3)
constexpr unsigned CTL_V = 0x00010000u | (1u << 20) | (5u << 22) | (3u << 25);

// ---------------------------------------------------------------------------
// Kernel 0a: x[B,C,N] f32  ->  xb[B,N,C] bf16  (LDS tiled transpose+convert)
// ---------------------------------------------------------------------------
__global__ __launch_bounds__(256)
void cvt_x_kernel(const float* __restrict__ x, __bf16* __restrict__ xb) {
  __shared__ float tile[32][33];
  const int tx = threadIdx.x & 31, ty = threadIdx.x >> 5;
  const int bx = blockIdx.x, by = blockIdx.y, b = blockIdx.z;
  #pragma unroll
  for (int i = 0; i < 4; ++i) {
    int c = by * 32 + ty + i * 8;
    int n = bx * 32 + tx;
    tile[ty + i * 8][tx] = x[((size_t)b * Cc + c) * Np + n];  // coalesced in n
  }
  __syncthreads();
  #pragma unroll
  for (int i = 0; i < 4; ++i) {
    int n = bx * 32 + ty + i * 8;
    int c = by * 32 + tx;
    xb[((size_t)b * Np + n) * Cc + c] = (__bf16)tile[tx][ty + i * 8]; // coalesced in c
  }
}

// ---------------------------------------------------------------------------
// Kernel 0b: convert the four 256x256 f32 weight matrices to bf16 (packed)
// ---------------------------------------------------------------------------
__global__ __launch_bounds__(256)
void cvt_w_kernel(const float* __restrict__ Wq, const float* __restrict__ Wk,
                  const float* __restrict__ Wv, const float* __restrict__ Wp,
                  __bf16* __restrict__ Wb) {
  int i = blockIdx.x * 256 + threadIdx.x;   // 0 .. 4*65536-1
  int m = i >> 16, r = i & 65535;
  const float* s = (m == 0) ? Wq : (m == 1) ? Wk : (m == 2) ? Wv : Wp;
  Wb[i] = (__bf16)s[r];
}

// ---------------------------------------------------------------------------
// Kernel 1: projections.  out[m][b, n, o] = sum_c W[m][o,c] * x[b,c,n] + bias
// One block = 64 pixels x 256 out-channels for one matrix m in {q,k,v,p}.
// ---------------------------------------------------------------------------
__global__ __launch_bounds__(256)
void qkvp_kernel(const __bf16* __restrict__ xb, const __bf16* __restrict__ Wb,
                 const float* __restrict__ bq, const float* __restrict__ bk,
                 const float* __restrict__ bv, const float* __restrict__ bp,
                 __bf16* __restrict__ qb, __bf16* __restrict__ kbuf,
                 __bf16* __restrict__ vbuf, float* __restrict__ out) {
  const int lane = threadIdx.x & 31;
  const int w    = threadIdx.x >> 5;
  const int lc   = lane & 15, half = lane >> 4;
  const int mt   = w & 3, nh = w >> 2;
  const int b    = blockIdx.y, mat = blockIdx.z;
  const int n0   = blockIdx.x * 64;

  const __bf16* Abase = xb + ((size_t)b * Np + n0 + mt * 16 + lc) * Cc;
  const __bf16* Wmat  = Wb + (size_t)mat * Cc * Cc;

  const v8f vzero = {0.f, 0.f, 0.f, 0.f, 0.f, 0.f, 0.f, 0.f};
  v8f acc[8];
  #pragma unroll
  for (int t = 0; t < 8; ++t) acc[t] = vzero;

  #pragma unroll
  for (int kk = 0; kk < Cc; kk += 32) {
    v16bf a = load_frag(Abase + kk + half * 16, Abase + kk + half * 16 + 8);
    #pragma unroll
    for (int t = 0; t < 8; ++t) {
      int o = nh * 128 + t * 16 + lc;
      const __bf16* wp = Wmat + (size_t)o * Cc + kk + half * 16;
      acc[t] = wmma_bf16(a, load_frag(wp, wp + 8), acc[t]);
    }
  }

  const float* biasPtr = (mat == 0) ? bq : (mat == 1) ? bk : (mat == 2) ? bv : bp;
  #pragma unroll
  for (int t = 0; t < 8; ++t) {
    int o = nh * 128 + t * 16 + lc;
    float bias = biasPtr[o];
    #pragma unroll
    for (int r = 0; r < 8; ++r) {
      int pixel = n0 + mt * 16 + r + 8 * half;   // C/D layout: M = r + 8*half
      float v = acc[t][r] + bias;
      if (mat == 0)      qb  [((size_t)b * Np + pixel) * Cc + o] = (__bf16)v;
      else if (mat == 1) kbuf[((size_t)b * Np + pixel) * Cc + o] = (__bf16)v;
      else if (mat == 2) vbuf[((size_t)b * Cc + o) * Np + pixel] = (__bf16)v;
      else               out [((size_t)b * Cc + o) * Np + pixel] = v;  // proj
    }
  }
}

// ---------------------------------------------------------------------------
// Kernel 2: flash attention, K AND V tiles TDM-double-buffered in LDS.
// Wave 0 DMAs the next K block, wave 1 the next V block; TENSORcnt + barrier
// publishes tiles to all 8 waves.  Steady state: no global loads in the loop.
// LDS: K 132KB + V 136KB + P strips 34KB = 302KB (< 320KB/WGP).
// ---------------------------------------------------------------------------
__global__ __launch_bounds__(256)
void flash_kernel(const __bf16* __restrict__ qb, const __bf16* __restrict__ kbuf,
                  const __bf16* __restrict__ vbuf, float* __restrict__ out) {
  __shared__ __align__(16) __bf16 kt[2][KB][KPAD];   // K: [key][ch], padded
  __shared__ __align__(16) __bf16 vt[2][Cc][VPAD];   // V: [ch][key], padded
  __shared__ __align__(16) __bf16 lds_p[8][16][136]; // per-wave P strip
  const int lane = threadIdx.x & 31;
  const int w    = threadIdx.x >> 5;
  const int lc   = lane & 15, half = lane >> 4;
  const int b    = blockIdx.y;
  const int q0   = blockIdx.x * 128 + w * 16;
  const int NBLK = Np / KB;

  // Q fragments (A-layout) held in registers for the whole kernel
  const __bf16* qrow = qb + ((size_t)b * Np + q0 + lc) * Cc;
  v16bf qf[8];
  #pragma unroll
  for (int s = 0; s < 8; ++s)
    qf[s] = load_frag(qrow + s * 32 + half * 16, qrow + s * 32 + half * 16 + 8);

  // Kick off the first K/V tile DMAs (TDM ignores EXEC; one instr per wave)
  if (w == 0)
    tdm_load_tile(kbuf + (size_t)b * Np * Cc,
                  (unsigned)(size_t)(void*)&kt[0][0][0],
                  CTL_K, Cc, Np, Cc, KB, Cc);
  if (w == 1)
    tdm_load_tile(vbuf + (size_t)b * Cc * Np,
                  (unsigned)(size_t)(void*)&vt[0][0][0],
                  CTL_V, Np, Cc, KB, Cc, Np);

  const v8f vzero = {0.f, 0.f, 0.f, 0.f, 0.f, 0.f, 0.f, 0.f};
  v8f oacc[16];
  #pragma unroll
  for (int t = 0; t < 16; ++t) oacc[t] = vzero;
  float m_i[8], l_i[8];
  #pragma unroll
  for (int r = 0; r < 8; ++r) { m_i[r] = -1e30f; l_i[r] = 0.f; }

  for (int jb = 0; jb < NBLK; ++jb) {
    const int key0 = jb * KB;

    // Tiles [jb&1] ready?  Issuing waves drain TENSORcnt, then all waves sync.
    if (w <= 1) __builtin_amdgcn_s_wait_tensorcnt((short)0);
    __syncthreads();
    // Overlap: start DMAs for the next K/V block into the other buffers.
    if ((jb + 1) < NBLK) {
      if (w == 0)
        tdm_load_tile(kbuf + ((size_t)b * Np + key0 + KB) * Cc,
                      (unsigned)(size_t)(void*)&kt[(jb + 1) & 1][0][0],
                      CTL_K, Cc, Np, Cc, KB, Cc);
      if (w == 1)
        tdm_load_tile(vbuf + (size_t)b * Cc * Np + key0 + KB,
                      (unsigned)(size_t)(void*)&vt[(jb + 1) & 1][0][0],
                      CTL_V, Np, Cc, KB, Cc, Np);
    }

    const __bf16(*ktc)[KPAD] = kt[jb & 1];
    const __bf16(*vtc)[VPAD] = vt[jb & 1];

    // ---- S = Q * K^T  (16 x 128, K-dim = 256), K fragments from LDS ----
    v8f sacc[8];
    #pragma unroll
    for (int t = 0; t < 8; ++t) sacc[t] = vzero;
    #pragma unroll
    for (int s = 0; s < 8; ++s) {
      const int kk = s * 32;
      #pragma unroll
      for (int t = 0; t < 8; ++t) {
        const __bf16* kp = &ktc[t * 16 + lc][kk + half * 16];
        sacc[t] = wmma_bf16(qf[s], load_frag(kp, kp + 8), sacc[t]);
      }
    }

    // ---- online softmax; row r of this lane-half = q0 + r + 8*half ----
    float corr[8];
    #pragma unroll
    for (int r = 0; r < 8; ++r) {
      float mx = -1e30f;
      #pragma unroll
      for (int t = 0; t < 8; ++t) { sacc[t][r] *= SCALE; mx = fmaxf(mx, sacc[t][r]); }
      mx = fmaxf(mx, __shfl_xor(mx, 1, 32));
      mx = fmaxf(mx, __shfl_xor(mx, 2, 32));
      mx = fmaxf(mx, __shfl_xor(mx, 4, 32));
      mx = fmaxf(mx, __shfl_xor(mx, 8, 32));
      float newm = fmaxf(m_i[r], mx);
      corr[r] = exp2f((m_i[r] - newm) * LOG2E);
      m_i[r] = newm;
      float sum = 0.f;
      #pragma unroll
      for (int t = 0; t < 8; ++t) {
        float p = exp2f((sacc[t][r] - newm) * LOG2E);
        sacc[t][r] = p;
        sum += p;
      }
      sum += __shfl_xor(sum, 1, 32);
      sum += __shfl_xor(sum, 2, 32);
      sum += __shfl_xor(sum, 4, 32);
      sum += __shfl_xor(sum, 8, 32);
      l_i[r] = l_i[r] * corr[r] + sum;
    }
    #pragma unroll
    for (int t = 0; t < 16; ++t)
      #pragma unroll
      for (int r = 0; r < 8; ++r) oacc[t][r] *= corr[r];

    // ---- re-layout P: C/D layout -> A layout via per-wave LDS strip ----
    #pragma unroll
    for (int t = 0; t < 8; ++t)
      #pragma unroll
      for (int r = 0; r < 8; ++r)
        lds_p[w][r + 8 * half][t * 16 + lc] = (__bf16)sacc[t][r];

    // ---- O += P * V   (V tile [ch][key] in LDS -> B fragments contiguous) --
    const __bf16* prow = &lds_p[w][lc][0];
    #pragma unroll
    for (int s2 = 0; s2 < 4; ++s2) {
      const int kk2 = s2 * 32;
      v16bf a = load_frag(prow + kk2 + half * 16, prow + kk2 + half * 16 + 8);
      #pragma unroll
      for (int ct = 0; ct < 16; ++ct) {
        const __bf16* vp = &vtc[ct * 16 + lc][kk2 + half * 16];
        oacc[ct] = wmma_bf16(a, load_frag(vp, vp + 8), oacc[ct]);
      }
    }
  }

  // ---- epilogue: out (holds proj) += O / l ----
  #pragma unroll
  for (int r = 0; r < 8; ++r) {
    float inv = 1.0f / l_i[r];
    int pixel = q0 + r + 8 * half;
    #pragma unroll
    for (int ct = 0; ct < 16; ++ct) {
      int ch = ct * 16 + lc;
      size_t idx = ((size_t)b * Cc + ch) * Np + pixel;
      out[idx] += oacc[ct][r] * inv;
    }
  }
}

// ---------------------------------------------------------------------------
extern "C" void kernel_launch(void* const* d_in, const int* in_sizes, int n_in,
                              void* d_out, int out_size, void* d_ws, size_t ws_size,
                              hipStream_t stream) {
  (void)in_sizes; (void)n_in; (void)out_size; (void)ws_size;
  const float* x  = (const float*)d_in[0];
  const float* Wq = (const float*)d_in[1];
  const float* bq = (const float*)d_in[2];
  const float* Wk = (const float*)d_in[3];
  const float* bk = (const float*)d_in[4];
  const float* Wv = (const float*)d_in[5];
  const float* bv = (const float*)d_in[6];
  const float* Wp = (const float*)d_in[7];
  const float* bp = (const float*)d_in[8];
  float* out = (float*)d_out;

  char* ws = (char*)d_ws;
  size_t off = 0;
  auto take = [&](size_t bytes) -> char* {
    char* p = ws + off;
    off += (bytes + 255) & ~(size_t)255;
    return p;
  };
  __bf16* xb   = (__bf16*)take((size_t)Bb * Np * Cc * 2);  // 16 MB
  __bf16* Wb   = (__bf16*)take((size_t)4 * Cc * Cc * 2);   // 512 KB
  __bf16* qb   = (__bf16*)take((size_t)Bb * Np * Cc * 2);  // 16 MB
  __bf16* kbuf = (__bf16*)take((size_t)Bb * Np * Cc * 2);  // 16 MB
  __bf16* vbuf = (__bf16*)take((size_t)Bb * Np * Cc * 2);  // 16 MB

  cvt_x_kernel<<<dim3(Np / 32, Cc / 32, Bb), 256, 0, stream>>>(x, xb);
  cvt_w_kernel<<<dim3(4 * Cc * Cc / 256), 256, 0, stream>>>(Wq, Wk, Wv, Wp, Wb);
  qkvp_kernel<<<dim3(Np / 64, Bb, 4), 256, 0, stream>>>(xb, Wb, bq, bk, bv, bp,
                                                        qb, kbuf, vbuf, out);
  flash_kernel<<<dim3(Np / 128, Bb), 256, 0, stream>>>(qb, kbuf, vbuf, out);
}
